// TransformerBlock_13005160972526
// MI455X (gfx1250) — matrix operand order repaired and checked by reading, not compiled, not run on hardware
//
#include <hip/hip_runtime.h>
#include <hip/hip_bf16.h>

// ---------------- problem constants ----------------
constexpr int kH   = 16;     // heads
constexpr int kD   = 1024;   // model dim
constexpr int kKD  = 64;     // head dim
constexpr int kMFF = 4096;   // ffn dim
constexpr int kMEM = 3;
constexpr int kS   = 512;
constexpr int kB   = 4;
constexpr int kL   = (kMEM + 1) * kS;   // 2048 keys
constexpr int kTOK = kS * kB;           // 2048 query tokens
constexpr int kMTOK = kL * kB;          // 8192 key/value tokens

typedef __attribute__((ext_vector_type(16))) _Float16 v16h;
typedef __attribute__((ext_vector_type(8)))  float    v8f;

__device__ __forceinline__ v8f wmma_f16(v16h a, v16h b, v8f c) {
  // D = A(16x32 f16) x B(32x16 f16) + C(16x16 f32)
  return __builtin_amdgcn_wmma_f32_16x16x32_f16(
      false, a, false, b, (short)0, c, false, false);
}

// Async 16B copy global -> LDS (GLOBAL_LOAD_ASYNC_TO_LDS_B128, ASYNCcnt).
// Generic pointer to __shared__ has the LDS byte offset in its low 32 bits
// (CDNA5 aperture rule).
__device__ __forceinline__ void async_ld16(const _Float16* g, _Float16* l) {
  uint32_t loff = (uint32_t)(uintptr_t)l;
  asm volatile("global_load_async_to_lds_b128 %0, %1, off"
               :: "v"(loff), "v"(g) : "memory");
}
__device__ __forceinline__ void wait_async0() {
  asm volatile("s_wait_asynccnt 0" ::: "memory");
}

// ---- fragment builders (wave32; this code's contiguous k order, same for A and B) ----
// A 16x32 f16: lane l -> M = l&15, group g = l>>4;
//   VGPR v half h: K = 16g + 2v + h
__device__ __forceinline__ v16h frag_a(const _Float16* __restrict__ p, int ld) {
  const int lane = threadIdx.x & 31;
  const int m = lane & 15, g = (lane >> 4) & 1;
  const _Float16* row = p + m * ld;
  v16h a;
#pragma unroll
  for (int v = 0; v < 8; ++v) {
    const int k = 16 * g + 2 * v;
    a[2 * v]     = row[k];
    a[2 * v + 1] = row[k + 1];
  }
  return a;
}

// B 32x16 f16, element (k,n) = p[n*ld + k]  (source stored N-major, K contiguous)
// lane l -> N = l&15, group g = l>>4 ; VGPR v half h: K = 16g + 2v + h
__device__ __forceinline__ v16h frag_b_nk(const _Float16* __restrict__ p, int ld) {
  const int lane = threadIdx.x & 31;
  const int n = lane & 15, g = (lane >> 4) & 1;
  const _Float16* row = p + n * ld;
  v16h b;
#pragma unroll
  for (int v = 0; v < 8; ++v) {
    const int k = 16 * g + 2 * v;
    b[2 * v]     = row[k];
    b[2 * v + 1] = row[k + 1];
  }
  return b;
}

// B 32x16 f16, element (k,n) = p[k*ld + n]  (source stored K-major)
__device__ __forceinline__ v16h frag_b_km(const _Float16* __restrict__ p, int ld) {
  const int lane = threadIdx.x & 31;
  const int n = lane & 15, g = (lane >> 4) & 1;
  v16h b;
#pragma unroll
  for (int v = 0; v < 8; ++v) {
    const int k = 16 * g + 2 * v;
    b[2 * v]     = p[k * ld + n];
    b[2 * v + 1] = p[(k + 1) * ld + n];
  }
  return b;
}

// ---------------- f32 -> f16 pack ----------------
__global__ void pack_half(const float* __restrict__ src, _Float16* __restrict__ dst, int n) {
  int i = blockIdx.x * blockDim.x + threadIdx.x;
  if (i < n) dst[i] = (_Float16)src[i];
}

// ---------------- generic tiled NT GEMM ----------------
// C[M,N] = scale * (A[M,K] x W[N,K]^T) + bias[N]; optional relu; f32/f16 outs.
// block = 128 threads (4 waves), tile 128x64 (wave -> 32 rows), K-step 32.
// Tiles staged with async global->LDS copies.
__global__ __launch_bounds__(128) void gemm_nt(
    const _Float16* __restrict__ A, const _Float16* __restrict__ W,
    const float* __restrict__ bias, float* __restrict__ C32,
    _Float16* __restrict__ C16, int M, int N, int K, float scale, int relu) {
  __shared__ _Float16 sA[128 * 32];
  __shared__ _Float16 sB[64 * 32];
  const int tid = threadIdx.x;
  const int w = tid >> 5, lane = tid & 31;
  const int bm = blockIdx.y * 128, bn = blockIdx.x * 64;

  v8f acc0[4] = {};  // rows w*32 .. w*32+15
  v8f acc1[4] = {};  // rows w*32+16 .. w*32+31

  for (int k0 = 0; k0 < K; k0 += 32) {
    __syncthreads();  // previous compute done before tiles are overwritten
    // stage A: 128x32 halfs = 512 x 16B units, 4 per thread
#pragma unroll
    for (int j = 0; j < 4; ++j) {
      const int u = j * 128 + tid;
      const int row = u >> 1, col = (u & 1) * 16;
      async_ld16(A + (size_t)(bm + row) * K + k0 + col, &sA[row * 32 + col]);
    }
    // stage W: 64x32 halfs = 256 x 16B units, 2 per thread
#pragma unroll
    for (int j = 0; j < 2; ++j) {
      const int u = j * 128 + tid;
      const int row = u >> 1, col = (u & 1) * 16;
      async_ld16(W + (size_t)(bn + row) * K + k0 + col, &sB[row * 32 + col]);
    }
    if (k0 + 32 < K) {  // keep next K tile warm in near caches
      __builtin_prefetch(A + (size_t)(bm + (tid >> 1)) * K + k0 + 32, 0, 3);
      __builtin_prefetch(W + (size_t)(bn + (tid & 63)) * K + k0 + 32, 0, 3);
    }
    wait_async0();
    __syncthreads();

    const v16h af0 = frag_a(sA + (w * 32) * 32, 32);
    const v16h af1 = frag_a(sA + (w * 32 + 16) * 32, 32);
#pragma unroll
    for (int nt = 0; nt < 4; ++nt) {
      const v16h bf = frag_b_nk(sB + nt * 16 * 32, 32);
      acc0[nt] = wmma_f16(af0, bf, acc0[nt]);
      acc1[nt] = wmma_f16(af1, bf, acc1[nt]);
    }
  }

  const int g = (lane >> 4) & 1, n0 = lane & 15;
#pragma unroll
  for (int half = 0; half < 2; ++half) {
    v8f* acc = half ? acc1 : acc0;
#pragma unroll
    for (int nt = 0; nt < 4; ++nt) {
#pragma unroll
      for (int r = 0; r < 8; ++r) {
        const int row = bm + w * 32 + half * 16 + r + 8 * g;
        const int col = bn + nt * 16 + n0;
        float v = acc[nt][r] * scale + (bias ? bias[col] : 0.f);
        if (relu) v = fmaxf(v, 0.f);
        if (C32) C32[(size_t)row * N + col] = v;
        if (C16) C16[(size_t)row * N + col] = (_Float16)v;
      }
    }
  }
}

// ---------------- flash attention ----------------
// One block per (head, batch, 16-row S tile). 4 waves split L=2048.
// Q is pre-scaled by 1/sqrt(kKD). Output AV is (S,B,H*KD) f16.
__global__ __launch_bounds__(128) void attention_kernel(
    const _Float16* __restrict__ Qh, const _Float16* __restrict__ Kh,
    const _Float16* __restrict__ Vh, const float* __restrict__ mask,
    _Float16* __restrict__ AV) {
  __shared__ _Float16 sQ[16 * 64];
  __shared__ _Float16 sK[4][32 * 64];
  __shared__ _Float16 sV[4][32 * 64];
  __shared__ _Float16 sP[4][16 * 32];
  __shared__ float sO[4][16 * 64];
  __shared__ float wmax[4][16], wsum[4][16];
  __shared__ float gmaxs[16], gsums[16];

  const int blk = blockIdx.x;
  const int st = blk & 31;             // S tile (32 tiles of 16)
  const int b  = (blk >> 5) & 3;       // batch
  const int hh = blk >> 7;             // head
  const int tid = threadIdx.x;
  const int w = tid >> 5, lane = tid & 31;
  const int g = (lane >> 4) & 1, n0 = lane & 15;

  // stage Q tile (16 x 64 halfs) cooperatively via async copy
  {
    const int m = tid >> 3, c = (tid & 7) * 8;
    const int srow = st * 16 + m;
    async_ld16(Qh + ((size_t)srow * kB + b) * kD + hh * kKD + c, &sQ[m * 64 + c]);
  }
  wait_async0();
  __syncthreads();
  const v16h qa0 = frag_a(sQ, 64);        // kdim 0..31
  const v16h qa1 = frag_a(sQ + 32, 64);   // kdim 32..63

  _Float16* sKw = sK[w];
  _Float16* sVw = sV[w];
  _Float16* sPw = sP[w];

  float rmax[8], rsum[8];
#pragma unroll
  for (int r = 0; r < 8; ++r) { rmax[r] = -1e30f; rsum[r] = 0.f; }
  v8f o[4] = {};

  for (int c = 0; c < 16; ++c) {          // 16 chunks of 32 keys per wave
    const int lbase = w * 512 + c * 32;
    // wave-cooperative async stage of K,V chunk (32 x 64 halfs each)
#pragma unroll
    for (int i = 0; i < 8; ++i) {
      const int idx = i * 32 + lane;
      const int row = idx >> 3, cc = (idx & 7) * 8;
      const size_t gk = ((size_t)(lbase + row) * kB + b) * kD + hh * kKD + cc;
      async_ld16(Kh + gk, &sKw[row * 64 + cc]);
      async_ld16(Vh + gk, &sVw[row * 64 + cc]);
    }
    wait_async0();  // wave-local: this wave's LDS region is now valid

    // scores: Q(16x64) x K^T(64x32) -> two 16x16 f32 tiles
    v8f sc[2];
#pragma unroll
    for (int lt = 0; lt < 2; ++lt) {
      v8f z = {};
      z = wmma_f16(qa0, frag_b_nk(sKw + lt * 16 * 64, 64), z);
      z = wmma_f16(qa1, frag_b_nk(sKw + lt * 16 * 64 + 32, 64), z);
      sc[lt] = z;
    }

    // online softmax (rows of this lane: m = r + 8g)
#pragma unroll
    for (int r = 0; r < 8; ++r) {
      const int m = r + 8 * g;
      const int srow = st * 16 + m;
      float s0 = sc[0][r] + mask[(size_t)srow * kL + lbase + n0];
      float s1 = sc[1][r] + mask[(size_t)srow * kL + lbase + 16 + n0];
      float cm = fmaxf(s0, s1);
#pragma unroll
      for (int d = 1; d < 16; d <<= 1) cm = fmaxf(cm, __shfl_xor(cm, d, 32));
      const float nm = fmaxf(rmax[r], cm);
      const float rescale = __expf(rmax[r] - nm);
      const float p0 = __expf(s0 - nm), p1 = __expf(s1 - nm);
      float ps = p0 + p1;
#pragma unroll
      for (int d = 1; d < 16; d <<= 1) ps += __shfl_xor(ps, d, 32);
      rsum[r] = rsum[r] * rescale + ps;
      rmax[r] = nm;
#pragma unroll
      for (int nt = 0; nt < 4; ++nt) o[nt][r] *= rescale;
      sPw[m * 32 + n0]      = (_Float16)p0;
      sPw[m * 32 + 16 + n0] = (_Float16)p1;
    }
    asm volatile("s_wait_dscnt 0" ::: "memory");  // P visible wave-locally

    // O += P(16x32) x V(32x64)
    const v16h pa = frag_a(sPw, 32);
#pragma unroll
    for (int nt = 0; nt < 4; ++nt)
      o[nt] = wmma_f16(pa, frag_b_km(sVw + nt * 16, 64), o[nt]);
  }

  // publish per-wave stats + partial O
  if (n0 == 0) {
#pragma unroll
    for (int r = 0; r < 8; ++r) {
      wmax[w][r + 8 * g] = rmax[r];
      wsum[w][r + 8 * g] = rsum[r];
    }
  }
#pragma unroll
  for (int nt = 0; nt < 4; ++nt)
#pragma unroll
    for (int r = 0; r < 8; ++r)
      sO[w][(r + 8 * g) * 64 + nt * 16 + n0] = o[nt][r];
  __syncthreads();

  if (tid < 16) {
    float gm = -1e30f;
    for (int ww = 0; ww < 4; ++ww) gm = fmaxf(gm, wmax[ww][tid]);
    float gs = 0.f;
    for (int ww = 0; ww < 4; ++ww) gs += wsum[ww][tid] * __expf(wmax[ww][tid] - gm);
    gmaxs[tid] = gm;
    gsums[tid] = gs;
  }
  __syncthreads();

  for (int idx = tid; idx < 16 * 64; idx += 128) {
    const int m = idx >> 6, kd = idx & 63;
    float accv = 0.f;
    for (int ww = 0; ww < 4; ++ww)
      accv += sO[ww][m * 64 + kd] * __expf(wmax[ww][m] - gmaxs[m]);
    accv /= gsums[m];
    AV[((size_t)(st * 16 + m) * kB + b) * kD + hh * kKD + kd] = (_Float16)accv;
  }
}

// ---------------- residual add + LayerNorm ----------------
__global__ __launch_bounds__(256) void add_ln(
    const float* __restrict__ in, const float* __restrict__ res,
    const float* __restrict__ gamma, const float* __restrict__ beta,
    float* __restrict__ out32, _Float16* __restrict__ out16) {
  const int t = blockIdx.x, tid = threadIdx.x;
  __shared__ float red[256];
  __shared__ float s_mean, s_rstd;
  const size_t base = (size_t)t * kD;
  float v[4];
  float s = 0.f;
#pragma unroll
  for (int i = 0; i < 4; ++i) {
    const int c = tid + i * 256;
    v[i] = in[base + c] + res[base + c];
    s += v[i];
  }
  red[tid] = s;
  __syncthreads();
  for (int d = 128; d > 0; d >>= 1) {
    if (tid < d) red[tid] += red[tid + d];
    __syncthreads();
  }
  if (tid == 0) s_mean = red[0] * (1.f / kD);
  __syncthreads();
  const float mean = s_mean;
  float q = 0.f;
#pragma unroll
  for (int i = 0; i < 4; ++i) { const float d = v[i] - mean; q += d * d; }
  red[tid] = q;
  __syncthreads();
  for (int d = 128; d > 0; d >>= 1) {
    if (tid < d) red[tid] += red[tid + d];
    __syncthreads();
  }
  if (tid == 0) s_rstd = rsqrtf(red[0] * (1.f / kD) + 1e-5f);
  __syncthreads();
  const float rstd = s_rstd;
#pragma unroll
  for (int i = 0; i < 4; ++i) {
    const int c = tid + i * 256;
    const float y = (v[i] - mean) * rstd * gamma[c] + beta[c];
    if (out32) out32[base + c] = y;
    if (out16) out16[base + c] = (_Float16)y;
  }
}

// ---------------- host launcher ----------------
extern "C" void kernel_launch(void* const* d_in, const int* in_sizes, int n_in,
                              void* d_out, int out_size, void* d_ws, size_t ws_size,
                              hipStream_t stream) {
  (void)in_sizes; (void)n_in; (void)out_size; (void)ws_size;
  const float* x    = (const float*)d_in[0];
  const float* mask = (const float*)d_in[1];
  const float* mem  = (const float*)d_in[2];
  // d_in[3] = heads (16, hardcoded)
  const float* wq = (const float*)d_in[4];
  const float* wk = (const float*)d_in[5];
  const float* wv = (const float*)d_in[6];
  const float* wc = (const float*)d_in[7];
  const float* w1 = (const float*)d_in[8];
  const float* b1 = (const float*)d_in[9];
  const float* w2 = (const float*)d_in[10];
  const float* b2 = (const float*)d_in[11];
  const float* ln1g = (const float*)d_in[12];
  const float* ln1b = (const float*)d_in[13];
  const float* ln2g = (const float*)d_in[14];
  const float* ln2b = (const float*)d_in[15];
  float* out = (float*)d_out;

  char* ws = (char*)d_ws;
  size_t off = 0;
  auto take = [&](size_t bytes) -> char* {
    char* p = ws + off;
    off = (off + bytes + 255) & ~(size_t)255;
    return p;
  };
  _Float16* XMh = (_Float16*)take((size_t)kMTOK * kD * 2);  // [memory; x] f16
  _Float16* Wqh = (_Float16*)take((size_t)kD * kD * 2);
  _Float16* Wkh = (_Float16*)take((size_t)kD * kD * 2);
  _Float16* Wvh = (_Float16*)take((size_t)kD * kD * 2);
  _Float16* Wch = (_Float16*)take((size_t)kD * kD * 2);
  _Float16* W1h = (_Float16*)take((size_t)kMFF * kD * 2);
  _Float16* W2h = (_Float16*)take((size_t)kD * kMFF * 2);
  _Float16* Qf  = (_Float16*)take((size_t)kTOK * kD * 2);
  _Float16* Kf  = (_Float16*)take((size_t)kMTOK * kD * 2);
  _Float16* Vf  = (_Float16*)take((size_t)kMTOK * kD * 2);
  _Float16* AVh = (_Float16*)take((size_t)kTOK * kD * 2);
  float*    U0  = (float*)take((size_t)kTOK * kD * 4);      // wc out; reused as FFN2 out
  float*    U1  = (float*)take((size_t)kTOK * kD * 4);      // post-LN1 f32
  _Float16* Uh  = (_Float16*)take((size_t)kTOK * kD * 2);   // post-LN1 f16
  _Float16* H1h = XMh;   // FFN hidden f16 (reuses XMh, dead after k/v GEMMs; same size)
  float*    Z0  = U0;    // FFN2 f32 out (reuses U0, dead after LN1)

  auto packn = [&](const float* s, _Float16* d, int n) {
    pack_half<<<(n + 255) / 256, 256, 0, stream>>>(s, d, n);
  };

  // 1) pack weights and [memory; x]
  packn(wq, Wqh, kD * kD);
  packn(wk, Wkh, kD * kD);
  packn(wv, Wvh, kD * kD);
  packn(wc, Wch, kD * kD);
  packn(w1, W1h, kMFF * kD);
  packn(w2, W2h, kD * kMFF);
  packn(mem, XMh, kMEM * kS * kB * kD);
  packn(x, XMh + (size_t)(kMEM * kS * kB) * kD, kS * kB * kD);
  const _Float16* Xh = XMh + (size_t)(kMEM * kS * kB) * kD;  // x f16 view

  // 2) projections (q pre-scaled by 1/sqrt(64) = 0.125)
  gemm_nt<<<dim3(kD / 64, kTOK / 128), 128, 0, stream>>>(
      Xh, Wqh, nullptr, nullptr, Qf, kTOK, kD, kD, 0.125f, 0);
  gemm_nt<<<dim3(kD / 64, kMTOK / 128), 128, 0, stream>>>(
      XMh, Wkh, nullptr, nullptr, Kf, kMTOK, kD, kD, 1.f, 0);
  gemm_nt<<<dim3(kD / 64, kMTOK / 128), 128, 0, stream>>>(
      XMh, Wvh, nullptr, nullptr, Vf, kMTOK, kD, kD, 1.f, 0);

  // 3) attention: 16 heads x 4 batch x 32 S-tiles
  attention_kernel<<<kH * kB * (kS / 16), 128, 0, stream>>>(Qf, Kf, Vf, mask, AVh);

  // 4) output projection + residual LN1
  gemm_nt<<<dim3(kD / 64, kTOK / 128), 128, 0, stream>>>(
      AVh, Wch, nullptr, U0, nullptr, kTOK, kD, kD, 1.f, 0);
  add_ln<<<kTOK, 256, 0, stream>>>(U0, x, ln1g, ln1b, U1, Uh);

  // 5) FFN + residual LN2 -> d_out
  gemm_nt<<<dim3(kMFF / 64, kTOK / 128), 128, 0, stream>>>(
      Uh, W1h, b1, nullptr, H1h, kTOK, kMFF, kD, 1.f, 1);
  gemm_nt<<<dim3(kD / 64, kTOK / 128), 128, 0, stream>>>(
      H1h, W2h, b2, Z0, nullptr, kTOK, kD, kMFF, 1.f, 0);
  add_ln<<<kTOK, 256, 0, stream>>>(Z0, U1, ln2g, ln2b, out, nullptr);
}
